// Renderer_super_89687507075984
// MI455X (gfx1250) — matrix operand order repaired and checked by reading, not compiled, not run on hardware
//
#include <hip/hip_runtime.h>

// ---------------------------------------------------------------------------
// CDNA5 (gfx1250) neural point-renderer, wave32 + v_wmma_f32_16x16x32_bf16,
// async LDS staging (global_load_async_to_lds_b128).
//   H=W=512, K=8, DIM=32, HID=128
// Workspace layout (assumes ws_size >= ~563 MB):
//   [0,         268435456)  fm   bf16 [K,H,W,64]  (fz ch0..31 | fc ch32..63)
//   [268435456, 536870912)  t1   bf16 [K,H,W,64]  (sup1 out, later mpn1 out)
//   [536870912, 545259520)  fuse f32  [H,W,K]
//   [545259520, 562036736)  u1   bf16 [H,W,32]
//   [562036736, ...)        transposed/padded bf16 weights (prep kernels)
// ---------------------------------------------------------------------------

#define Hh 512
#define Ww 512
#define KK 8

typedef __attribute__((ext_vector_type(16))) __bf16 v16bf;
typedef __attribute__((ext_vector_type(8)))  float  v8f;

union Frag { v16bf v; uint4 u[2]; };

__device__ __forceinline__ v8f wmma_bf16(v16bf a, v16bf b, v8f c) {
  return __builtin_amdgcn_wmma_f32_16x16x32_bf16(false, a, false, b, (short)0, c,
                                                 false, false);
}

// ---------------------------------------------------------------------------
// Weight prep: transpose to [Npad][Kpad] bf16 with zero padding so B fragments
// are two 16-byte runs per lane, 32 bytes apart (branch-free global_load_b128 pairs).
// ---------------------------------------------------------------------------
__global__ __launch_bounds__(256) void prep_dense(const float* __restrict__ W,
                                                  __bf16* __restrict__ Wt,
                                                  int Kdim, int N, int KPAD) {
  int idx = blockIdx.x * 256 + threadIdx.x;
  if (idx >= N * KPAD) return;
  int n = idx / KPAD, k = idx - n * KPAD;
  __bf16 v = (__bf16)0.f;
  if (k < Kdim) v = (__bf16)W[(size_t)k * N + n];
  Wt[idx] = v;
}

__global__ __launch_bounds__(256) void prep_conv(const float* __restrict__ W,
                                                 __bf16* __restrict__ Wt,
                                                 int CIN, int CP, int COUT,
                                                 int NPAD, int KPAD) {
  int idx = blockIdx.x * 256 + threadIdx.x;
  if (idx >= NPAD * KPAD) return;
  int n = idx / KPAD, k = idx - n * KPAD;
  int kp = k / CP, ci = k - kp * CP;
  __bf16 v = (__bf16)0.f;
  if (kp < 9 && ci < CIN && n < COUT)
    v = (__bf16)W[(size_t)(kp * CIN + ci) * COUT + n];
  Wt[idx] = v;
}

// ---------------------------------------------------------------------------
// MLP: one wave = 16 points. Activations in per-wave LDS rows (stride 168,
// 16B-aligned, zero-padded). A = 2x ds_load_b128, B = 2x global_load_b128.
// ---------------------------------------------------------------------------
#define ACTS 168

template <int KPAD, int KC, int NT>
__device__ __forceinline__ void dense_layer(const __bf16* act,
                                            const __bf16* __restrict__ Wt,
                                            int lane, v8f* acc) {
  const int grp = lane >> 4, m = lane & 15;
  const __bf16* arow = act + m * ACTS;
#pragma unroll
  for (int kc = 0; kc < KC; ++kc) {
    Frag a;
    int k0 = kc * 32 + grp * 8;
    a.u[0] = *(const uint4*)(arow + k0);
    a.u[1] = *(const uint4*)(arow + k0 + 16);
#pragma unroll
    for (int nt = 0; nt < NT; ++nt) {
      Frag b;
      const __bf16* bp = Wt + (size_t)(nt * 16 + m) * KPAD + kc * 32 + grp * 8;
      b.u[0] = *(const uint4*)bp;
      b.u[1] = *(const uint4*)(bp + 16);
      acc[nt] = wmma_bf16(a.v, b.v, acc[nt]);
    }
  }
}

template <int NT>
__device__ __forceinline__ void epi_relu(const v8f* acc,
                                         const float* __restrict__ bias,
                                         __bf16* act, int lane) {
  const int grp = lane >> 4, m = lane & 15;
#pragma unroll
  for (int nt = 0; nt < NT; ++nt) {
    float bb = bias[nt * 16 + m];
#pragma unroll
    for (int r = 0; r < 8; ++r) {
      float v = acc[nt][r] + bb;
      act[(r + grp * 8) * ACTS + nt * 16 + m] = (__bf16)fmaxf(v, 0.f);
    }
  }
}

__global__ __launch_bounds__(128) void mlp_kernel(
    const float* __restrict__ ray, const float* __restrict__ zbufs,
    const __bf16* __restrict__ W1t, const float* __restrict__ b1,
    const __bf16* __restrict__ W2t, const float* __restrict__ b2,
    const __bf16* __restrict__ W3t, const float* __restrict__ b3,
    const __bf16* __restrict__ W4t, const float* __restrict__ b4,
    __bf16* __restrict__ fm) {
  __shared__ __bf16 sAct[4][16][ACTS];
  __shared__ float  sMask[4][16];

  const int lane = threadIdx.x & 31, wave = threadIdx.x >> 5;
  const int grp = lane >> 4, m = lane & 15;
  const size_t p0 = ((size_t)blockIdx.x * 4 + wave) * 16;
  __bf16* act = &sAct[wave][0][0];

  for (int t = lane; t < 16 * ACTS / 8; t += 32) {
    uint4 z; z.x = z.y = z.z = z.w = 0u;
    *(uint4*)(act + t * 8) = z;
  }
  if (lane < 16) {
    size_t p = p0 + lane;
    int kk  = (int)(p / (Hh * Ww));
    int rem = (int)(p % (Hh * Ww));
    const float* r7 = ray + (size_t)rem * 7;
    float z = zbufs[(size_t)rem * KK + kk];
    float s = z / r7[6];
#pragma unroll
    for (int j = 0; j < 3; ++j) {
      sAct[wave][lane][j]       = (__bf16)(r7[j] + r7[3 + j] * s);
      sAct[wave][lane][128 + j] = (__bf16)r7[3 + j];
    }
    sMask[wave][lane] = (z > 0.f) ? 1.f : 0.f;
  }

  {  // layer 1: 3 -> 128 relu (Kpad 32)
    v8f acc[8];
#pragma unroll
    for (int nt = 0; nt < 8; ++nt) acc[nt] = {0, 0, 0, 0, 0, 0, 0, 0};
    dense_layer<32, 1, 8>(act, W1t, lane, acc);
    epi_relu<8>(acc, b1, act, lane);
  }
  {  // layer 2: 128 -> 128 relu
    v8f acc[8];
#pragma unroll
    for (int nt = 0; nt < 8; ++nt) acc[nt] = {0, 0, 0, 0, 0, 0, 0, 0};
    dense_layer<128, 4, 8>(act, W2t, lane, acc);
    epi_relu<8>(acc, b2, act, lane);
  }
  {  // layer 3: cat(h,d) 131 -> 128 relu (Kpad 160)
    v8f acc[8];
#pragma unroll
    for (int nt = 0; nt < 8; ++nt) acc[nt] = {0, 0, 0, 0, 0, 0, 0, 0};
    dense_layer<160, 5, 8>(act, W3t, lane, acc);
    epi_relu<8>(acc, b3, act, lane);
  }
  {  // layer 4: 128 -> 32, +bias, *mask, store fm[...,0:32]
    v8f acc[2];
#pragma unroll
    for (int nt = 0; nt < 2; ++nt) acc[nt] = {0, 0, 0, 0, 0, 0, 0, 0};
    dense_layer<128, 4, 2>(act, W4t, lane, acc);
#pragma unroll
    for (int nt = 0; nt < 2; ++nt) {
      float bb = b4[nt * 16 + m];
#pragma unroll
      for (int r = 0; r < 8; ++r) {
        int mm = r + grp * 8;
        float v = (acc[nt][r] + bb) * sMask[wave][mm];
        fm[(p0 + mm) * 64 + nt * 16 + m] = (__bf16)v;
      }
    }
  }
}

// ---------------------------------------------------------------------------
// Implicit-GEMM 3x3 SAME conv, block-wide 64-pixel strip.
// Shared LDS halo tile (4 rows x 66 cols x CP; row 3 zeroed absorbs K-padding
// overrun). Aligned bf16 inputs staged with GLOBAL_LOAD_ASYNC_TO_LDS_B128
// (ASYNCcnt); OOB / zero-row chunks use plain zero ds_stores.
// ---------------------------------------------------------------------------
template <int CIN, int CP, bool IN_F32>
__device__ __forceinline__ void conv_stage_block(const void* __restrict__ inp,
                                                 int in_stride, int chan_off,
                                                 int in_batched, int kk, int y,
                                                 int xb, int tid, __bf16* sIn) {
  if constexpr (!IN_F32 && CIN == CP) {
    constexpr int CH8 = CP / 8;
    constexpr int TOT8 = 4 * 66 * CH8;
    for (int t = tid; t < TOT8; t += 128) {
      int c8 = t % CH8, xx = (t / CH8) % 66, ry = t / (CH8 * 66);
      int gy = y + ry - 1, gx = xb + xx - 1;
      __bf16* dst = sIn + t * 8;
      if (ry < 3 && (unsigned)gy < Hh && (unsigned)gx < Ww) {
        size_t pix = in_batched ? (((size_t)kk * Hh + gy) * Ww + gx)
                                : ((size_t)gy * Ww + gx);
        const __bf16* src =
            (const __bf16*)inp + pix * in_stride + chan_off + c8 * 8;
        unsigned ldsoff = (unsigned)(uintptr_t)dst;  // LDS addr = addr[31:0]
        asm volatile("global_load_async_to_lds_b128 %0, %1, off"
                     :: "v"(ldsoff), "v"(src) : "memory");
      } else {
        uint4 z; z.x = z.y = z.z = z.w = 0u;
        *(uint4*)dst = z;
      }
    }
    asm volatile("s_wait_asynccnt 0" ::: "memory");
  } else {
    constexpr int TOT = 4 * 66 * CP;
    for (int t = tid; t < TOT; t += 128) {
      int ci = t % CP, xx = (t / CP) % 66, ry = t / (CP * 66);
      __bf16 v = (__bf16)0.f;
      int gy = y + ry - 1, gx = xb + xx - 1;
      if (ry < 3 && ci < CIN && (unsigned)gy < Hh && (unsigned)gx < Ww) {
        size_t pix = in_batched ? (((size_t)kk * Hh + gy) * Ww + gx)
                                : ((size_t)gy * Ww + gx);
        size_t id = pix * in_stride + chan_off + ci;
        if constexpr (IN_F32) v = (__bf16)((const float*)inp)[id];
        else                  v = ((const __bf16*)inp)[id];
      }
      sIn[t] = v;
    }
  }
}

template <int CP, int KC, int KPAD, int MT>
__device__ __forceinline__ void conv_mma_strip(const __bf16* sIn,
                                               const __bf16* __restrict__ Wt,
                                               int lane, int nt, int tile0,
                                               v8f* acc) {
  const int grp = lane >> 4, m = lane & 15;
#pragma unroll
  for (int kc = 0; kc < KC; ++kc) {
    Frag b;
    const __bf16* bp = Wt + (size_t)(nt * 16 + m) * KPAD + kc * 32 + grp * 8;
    b.u[0] = *(const uint4*)bp;
    b.u[1] = *(const uint4*)(bp + 16);
    int gk0 = kc * 32 + grp * 8, gk1 = gk0 + 16;
    int kp0 = gk0 / CP, c0 = gk0 - kp0 * CP;
    int kp1 = gk1 / CP, c1 = gk1 - kp1 * CP;
    int base0 = ((kp0 / 3) * 66 + kp0 % 3) * CP + c0;
    int base1 = ((kp1 / 3) * 66 + kp1 % 3) * CP + c1;
#pragma unroll
    for (int t = 0; t < MT; ++t) {
      int col = (tile0 + t) * 16 + m;
      Frag a;
      a.u[0] = *(const uint4*)(sIn + base0 + col * CP);
      a.u[1] = *(const uint4*)(sIn + base1 + col * CP);
      acc[t] = wmma_bf16(a.v, b.v, acc[t]);
    }
  }
}

template <int CIN, int COUT, bool IN_F32, bool OUT_F32, bool RELU>
__global__ __launch_bounds__(128) void conv3x3_kernel(
    const void* __restrict__ inp, int in_stride, int in_koff, int in_batched,
    const __bf16* __restrict__ Wt, const float* __restrict__ bias,
    void* __restrict__ outp, int out_stride, int out_off, int batch) {
  constexpr int CP   = (CIN < 8) ? 8 : CIN;
  constexpr int KC   = (9 * CP + 31) / 32;
  constexpr int KPAD = KC * 32;
  constexpr int NT   = (COUT + 15) / 16;
  constexpr int MT   = (NT >= 4) ? 4 : (NT == 2 ? 2 : 1);
  __shared__ __bf16 sIn[4 * 66 * CP];

  const int tid = threadIdx.x, lane = tid & 31, wave = tid >> 5;
  const int spr = Ww / 64;  // 8 strips per row
  int b   = blockIdx.x;
  int kk  = b / (Hh * spr);
  int rem = b % (Hh * spr);
  int y   = rem / spr;
  int xb  = (rem % spr) * 64;

  int chan_off = in_batched ? in_koff : in_koff * kk;
  conv_stage_block<CIN, CP, IN_F32>(inp, in_stride, chan_off, in_batched, kk, y,
                                    xb, tid, sIn);
  __syncthreads();

  int nt, tile0;
  if constexpr (NT >= 4)      { nt = wave;   tile0 = 0; }
  else if constexpr (NT == 2) { nt = wave & 1; tile0 = (wave >> 1) * 2; }
  else                        { nt = 0;      tile0 = wave; }

  v8f acc[MT];
#pragma unroll
  for (int t = 0; t < MT; ++t) acc[t] = {0, 0, 0, 0, 0, 0, 0, 0};
  conv_mma_strip<CP, KC, KPAD, MT>(sIn, Wt, lane, nt, tile0, acc);

  const int grp = lane >> 4, m = lane & 15;
  int co = nt * 16 + m;
#pragma unroll
  for (int t = 0; t < MT; ++t)
#pragma unroll
    for (int r = 0; r < 8; ++r) {
      if (co < COUT) {
        float v = acc[t][r] + bias[co];
        if (RELU) v = fmaxf(v, 0.f);
        int px = xb + (tile0 + t) * 16 + r + grp * 8;
        size_t pix = (batch > 1) ? (((size_t)kk * Hh + y) * Ww + px)
                                 : ((size_t)y * Ww + px);
        size_t idx = pix * out_stride + out_off + co;
        if (OUT_F32) ((float*)outp)[idx] = v;
        else         ((__bf16*)outp)[idx] = (__bf16)v;
      }
    }
}

// ---------------------------------------------------------------------------
// Fused mpn2 conv (64->64) + sigmoid + NeRF cumprod composite -> fuse[H,W,K]
// ---------------------------------------------------------------------------
__global__ __launch_bounds__(128) void mpn2_composite_kernel(
    const __bf16* __restrict__ t1, const __bf16* __restrict__ fm,
    const __bf16* __restrict__ Wt, const float* __restrict__ bias,
    float* __restrict__ fuse) {
  __shared__ __bf16 sIn[4 * 66 * 64];
  __shared__ float  sM[64][65];
  const int tid = threadIdx.x, lane = tid & 31, wave = tid >> 5;
  const int spr = Ww / 64;
  int b   = blockIdx.x;
  int kk  = b / (Hh * spr);
  int rem = b % (Hh * spr);
  int y   = rem / spr;
  int xb  = (rem % spr) * 64;

  conv_stage_block<64, 64, false>(t1, 64, 0, 1, kk, y, xb, tid, sIn);
  __syncthreads();

  const int nt = wave;
  v8f acc[4];
#pragma unroll
  for (int t = 0; t < 4; ++t) acc[t] = {0, 0, 0, 0, 0, 0, 0, 0};
  conv_mma_strip<64, 18, 576, 4>(sIn, Wt, lane, nt, 0, acc);

  const int grp = lane >> 4, m = lane & 15;
  float bb = bias[nt * 16 + m];
#pragma unroll
  for (int t = 0; t < 4; ++t)
#pragma unroll
    for (int r = 0; r < 8; ++r) {
      float v = acc[t][r] + bb;
      sM[t * 16 + r + grp * 8][nt * 16 + m] = 1.f / (1.f + __expf(-v));
    }
  __syncthreads();

  if (tid < 64) {
    int px = xb + tid;
    size_t pix = ((size_t)kk * Hh + y) * Ww + px;
    const __bf16* f = fm + pix * 64;
    float alpha = 1.f, s = 0.f;
    for (int c = 0; c < 64; ++c) {
      float mval = sM[tid][c];
      s += (float)f[c] * mval * alpha;
      alpha *= (1.f - mval);
    }
    fuse[((size_t)y * Ww + px) * KK + kk] = s;
  }
}

// ---------------------------------------------------------------------------

extern "C" void kernel_launch(void* const* d_in, const int* in_sizes, int n_in,
                              void* d_out, int out_size, void* d_ws,
                              size_t ws_size, hipStream_t stream) {
  const float* colors = (const float*)d_in[0];
  const float* ray    = (const float*)d_in[1];
  const float* zbufs  = (const float*)d_in[2];
  const float* W1 = (const float*)d_in[3];   const float* b1 = (const float*)d_in[4];
  const float* W2 = (const float*)d_in[5];   const float* b2 = (const float*)d_in[6];
  const float* W3 = (const float*)d_in[7];   const float* b3 = (const float*)d_in[8];
  const float* W4 = (const float*)d_in[9];   const float* b4 = (const float*)d_in[10];
  const float* k_sup1 = (const float*)d_in[11]; const float* b_sup1 = (const float*)d_in[12];
  const float* k_sup2 = (const float*)d_in[13]; const float* b_sup2 = (const float*)d_in[14];
  const float* k_mpn1 = (const float*)d_in[15]; const float* b_mpn1 = (const float*)d_in[16];
  const float* k_mpn2 = (const float*)d_in[17]; const float* b_mpn2 = (const float*)d_in[18];
  const float* k_un1  = (const float*)d_in[19]; const float* b_un1  = (const float*)d_in[20];
  const float* k_un2  = (const float*)d_in[21]; const float* b_un2  = (const float*)d_in[22];

  char* ws = (char*)d_ws;
  __bf16* fm   = (__bf16*)ws;
  __bf16* t1   = (__bf16*)(ws + 268435456ull);
  float*  fuse = (float*)(ws + 536870912ull);
  __bf16* u1   = (__bf16*)(ws + 545259520ull);

  size_t wo = 562036736ull;
  __bf16* W1t = (__bf16*)(ws + wo); wo += 128 * 32 * 2;    // Kpad 32
  __bf16* W2t = (__bf16*)(ws + wo); wo += 128 * 128 * 2;   // Kpad 128
  __bf16* W3t = (__bf16*)(ws + wo); wo += 128 * 160 * 2;   // Kpad 160
  __bf16* W4t = (__bf16*)(ws + wo); wo += 32 * 128 * 2;    // Kpad 128
  __bf16* sup1t = (__bf16*)(ws + wo); wo += 32 * 96 * 2;   // CP8  Kpad 96
  __bf16* sup2t = (__bf16*)(ws + wo); wo += 32 * 288 * 2;  // CP32 Kpad 288
  __bf16* mpn1t = (__bf16*)(ws + wo); wo += 64 * 576 * 2;  // CP64 Kpad 576
  __bf16* mpn2t = (__bf16*)(ws + wo); wo += 64 * 576 * 2;
  __bf16* un1t  = (__bf16*)(ws + wo); wo += 32 * 96 * 2;   // CP8  Kpad 96
  __bf16* un2t  = (__bf16*)(ws + wo); wo += 16 * 288 * 2;  // CP32 Kpad 288

  dim3 blk(128), pblk(256);
  // ---- weight prep ----
  prep_dense<<<16, pblk, 0, stream>>>(W1, W1t, 3, 128, 32);
  prep_dense<<<64, pblk, 0, stream>>>(W2, W2t, 128, 128, 128);
  prep_dense<<<80, pblk, 0, stream>>>(W3, W3t, 131, 128, 160);
  prep_dense<<<16, pblk, 0, stream>>>(W4, W4t, 128, 32, 128);
  prep_conv<<<12, pblk, 0, stream>>>(k_sup1, sup1t, 3, 8, 32, 32, 96);
  prep_conv<<<36, pblk, 0, stream>>>(k_sup2, sup2t, 32, 32, 32, 32, 288);
  prep_conv<<<144, pblk, 0, stream>>>(k_mpn1, mpn1t, 64, 64, 64, 64, 576);
  prep_conv<<<144, pblk, 0, stream>>>(k_mpn2, mpn2t, 64, 64, 64, 64, 576);
  prep_conv<<<12, pblk, 0, stream>>>(k_un1, un1t, 8, 8, 32, 32, 96);
  prep_conv<<<18, pblk, 0, stream>>>(k_un2, un2t, 32, 32, 3, 16, 288);

  // ---- pipeline ----
  mlp_kernel<<<32768, blk, 0, stream>>>(ray, zbufs, W1t, b1, W2t, b2, W3t, b3,
                                        W4t, b4, fm);
  conv3x3_kernel<3, 32, true, false, true><<<32768, blk, 0, stream>>>(
      colors, 24, 3, 0, sup1t, b_sup1, t1, 32, 0, 8);
  conv3x3_kernel<32, 32, false, false, false><<<32768, blk, 0, stream>>>(
      t1, 32, 0, 1, sup2t, b_sup2, fm, 64, 32, 8);
  conv3x3_kernel<64, 64, false, false, true><<<32768, blk, 0, stream>>>(
      fm, 64, 0, 1, mpn1t, b_mpn1, t1, 64, 0, 8);
  mpn2_composite_kernel<<<32768, blk, 0, stream>>>(t1, fm, mpn2t, b_mpn2, fuse);
  conv3x3_kernel<8, 32, true, false, true><<<4096, blk, 0, stream>>>(
      fuse, 8, 0, 0, un1t, b_un1, u1, 32, 0, 1);
  conv3x3_kernel<32, 3, false, true, false><<<4096, blk, 0, stream>>>(
      u1, 32, 0, 0, un2t, b_un2, d_out, 3, 0, 1);
}
